// GraphEncoder_91044716740861
// MI455X (gfx1250) — compile-verified, matrix operand order checked by reading, not run on hardware
//
#include <hip/hip_runtime.h>
#include <hip/hip_bf16.h>

// ---------------- problem constants (match reference) ----------------
#define NN   100000
#define EE   1600000
#define BB   64
#define DIN  64
#define DH   128
#define KITER 16
#define ALPHA 0.1f
#define LNEPS 1e-5f

typedef __attribute__((ext_vector_type(16))) _Float16 v16h;
typedef __attribute__((ext_vector_type(8)))  float    v8f;

__device__ __forceinline__ float gelu_f(float x) {
    return 0.5f * x * (1.0f + erff(x * 0.70710678118654752440f));
}
// order-preserving float -> uint encoding for atomicMax-based segment max
__device__ __forceinline__ unsigned fenc(float f) {
    unsigned u = __float_as_uint(f);
    return (u & 0x80000000u) ? ~u : (u | 0x80000000u);
}
__device__ __forceinline__ float fdec(unsigned e) {
    return __uint_as_float((e & 0x80000000u) ? (e ^ 0x80000000u) : ~e);
}

// ---------------- utility ----------------
__global__ void k_zero(float* p, int n) {
    for (int i = blockIdx.x * blockDim.x + threadIdx.x; i < n;
         i += gridDim.x * blockDim.x) p[i] = 0.0f;
}

// ---------------- weight pre-pack: f32 row-major -> f16 WMMA-B fragment order
// layout: half index = (((f*8)+nt)*32 + lane)*16 + i
// element = W[f*32 + kmap(lane,i)][nt*16 + (lane&15)], kmap: kh=(lane>>4)*8,
//           k = i<8 ? kh+i : 16+kh+(i-8)
template<int KF>
__global__ void k_pack(const float* __restrict__ W, _Float16* __restrict__ out) {
    int t = blockIdx.x * blockDim.x + threadIdx.x;
    if (t >= KF * 8 * 512) return;
    int i    = t & 15;
    int lane = (t >> 4) & 31;
    int nt   = (t >> 9) & 7;
    int f    = t >> 12;
    int kh   = (lane >> 4) * 8;
    int k    = f * 32 + (i < 8 ? kh + i : 16 + kh + (i - 8));
    int n    = nt * 16 + (lane & 15);
    out[t] = (_Float16)W[k * DH + n];
}

// ---------------- degree / gcn_norm precompute ----------------
__global__ void k_deg(const int* __restrict__ ei, float* deg) {
    int t = blockIdx.x * blockDim.x + threadIdx.x;
    if (t >= EE) return;
    atomicAdd(&deg[ei[EE + t]], 1.0f);
}
__global__ void k_degfin(float* dis, float* selfw) {
    int i = blockIdx.x * blockDim.x + threadIdx.x;
    if (i >= NN) return;
    float d = dis[i] + 1.0f;          // self loop
    selfw[i] = 1.0f / d;
    dis[i]   = rsqrtf(d);
}

// ---------------- GINE message + scatter: agg[dst] += relu(x[src]+ea@We+be)
template<int D>
__global__ void k_gine_edge(const int* __restrict__ ei, const float* __restrict__ x,
                            const float* __restrict__ ea, const float* __restrict__ We,
                            const float* __restrict__ be, float* agg) {
    constexpr int CH = D / 4;
    int t = blockIdx.x * blockDim.x + threadIdx.x;
    if (t >= EE * CH) return;
    int e = t / CH;
    int c = (t % CH) * 4;
    int s = ei[e], d = ei[EE + e];
    float a0 = ea[2 * e], a1 = ea[2 * e + 1];
    float4 xv = *(const float4*)(x + (size_t)s * D + c);
    float4 w0 = *(const float4*)(We + c);
    float4 w1 = *(const float4*)(We + D + c);
    float4 bv = *(const float4*)(be + c);
    float m0 = fmaxf(xv.x + a0 * w0.x + a1 * w1.x + bv.x, 0.0f);
    float m1 = fmaxf(xv.y + a0 * w0.y + a1 * w1.y + bv.y, 0.0f);
    float m2 = fmaxf(xv.z + a0 * w0.z + a1 * w1.z + bv.z, 0.0f);
    float m3 = fmaxf(xv.w + a0 * w0.w + a1 * w1.w + bv.w, 0.0f);
    float* ap = agg + (size_t)d * D + c;
    atomicAdd(ap + 0, m0); atomicAdd(ap + 1, m1);
    atomicAdd(ap + 2, m2); atomicAdd(ap + 3, m3);
}

// ---------------- fused MLP + LayerNorm, WMMA f16->f32, one wave = 16 rows --
// A = in1 + in2 (element-wise), then gelu(A@Wa+ba)@Wb+bb, optional gelu, LN.
template<int KIN, bool GEL2>
__global__ __launch_bounds__(32) void k_mlp(
    const float* __restrict__ in1, const float* __restrict__ in2,
    const _Float16* __restrict__ Wa, const float* __restrict__ ba,
    const _Float16* __restrict__ Wb, const float* __restrict__ bb,
    const float* __restrict__ gma, const float* __restrict__ bet,
    float* out1, float* out2) {
    __shared__ float smem[16 * DH];
    const int lane = threadIdx.x;
    const int m    = lane & 15;
    const int kh   = (lane >> 4) * 8;
    const int row  = blockIdx.x * 16 + m;     // 6250*16 == NN exactly

    // ---- build A fragments for GEMM1 (vectorized f32 loads, cvt to f16) ----
    v16h afr[KIN / 32];
    {
        const float* r1 = in1 + (size_t)row * KIN;
        const float* r2 = in2 + (size_t)row * KIN;
        for (int f = 0; f < KIN / 32; ++f) {
            float4 p[4], q[4];
            p[0] = *(const float4*)(r1 + f * 32 + kh);
            p[1] = *(const float4*)(r1 + f * 32 + kh + 4);
            p[2] = *(const float4*)(r1 + f * 32 + 16 + kh);
            p[3] = *(const float4*)(r1 + f * 32 + 16 + kh + 4);
            q[0] = *(const float4*)(r2 + f * 32 + kh);
            q[1] = *(const float4*)(r2 + f * 32 + kh + 4);
            q[2] = *(const float4*)(r2 + f * 32 + 16 + kh);
            q[3] = *(const float4*)(r2 + f * 32 + 16 + kh + 4);
#pragma unroll
            for (int g = 0; g < 4; ++g) {
                afr[f][g * 4 + 0] = (_Float16)(p[g].x + q[g].x);
                afr[f][g * 4 + 1] = (_Float16)(p[g].y + q[g].y);
                afr[f][g * 4 + 2] = (_Float16)(p[g].z + q[g].z);
                afr[f][g * 4 + 3] = (_Float16)(p[g].w + q[g].w);
            }
        }
    }
    // ---- GEMM1: (16xKIN)@(KINx128), bias + GELU, stash in LDS ----
    for (int nt = 0; nt < 8; ++nt) {
        v8f c = {0.f, 0.f, 0.f, 0.f, 0.f, 0.f, 0.f, 0.f};
        const int n = nt * 16 + (lane & 15);
        for (int f = 0; f < KIN / 32; ++f) {
            const v16h bfr = *(const v16h*)(Wa + ((((size_t)f * 8 + nt) * 32 + lane) << 4));
            c = __builtin_amdgcn_wmma_f32_16x16x32_f16(false, afr[f], false, bfr,
                                                       (short)0, c, false, false);
        }
        float bia = ba[n];
#pragma unroll
        for (int j = 0; j < 8; ++j) {
            int rr = j + ((lane >> 4) << 3);
            smem[rr * DH + n] = gelu_f(c[j] + bia);
        }
    }
    __syncthreads();
    // ---- re-layout LDS intermediate into A fragments for GEMM2 ----
    v16h a2[4];
    for (int f = 0; f < 4; ++f) {
        const float* sr = smem + m * DH + f * 32;
        float4 p0 = *(const float4*)(sr + kh);
        float4 p1 = *(const float4*)(sr + kh + 4);
        float4 p2 = *(const float4*)(sr + 16 + kh);
        float4 p3 = *(const float4*)(sr + 16 + kh + 4);
        a2[f][0]=(_Float16)p0.x;  a2[f][1]=(_Float16)p0.y;
        a2[f][2]=(_Float16)p0.z;  a2[f][3]=(_Float16)p0.w;
        a2[f][4]=(_Float16)p1.x;  a2[f][5]=(_Float16)p1.y;
        a2[f][6]=(_Float16)p1.z;  a2[f][7]=(_Float16)p1.w;
        a2[f][8]=(_Float16)p2.x;  a2[f][9]=(_Float16)p2.y;
        a2[f][10]=(_Float16)p2.z; a2[f][11]=(_Float16)p2.w;
        a2[f][12]=(_Float16)p3.x; a2[f][13]=(_Float16)p3.y;
        a2[f][14]=(_Float16)p3.z; a2[f][15]=(_Float16)p3.w;
    }
    __syncthreads();
    // ---- GEMM2: (16x128)@(128x128), bias (+gelu), back to LDS ----
    for (int nt = 0; nt < 8; ++nt) {
        v8f c = {0.f, 0.f, 0.f, 0.f, 0.f, 0.f, 0.f, 0.f};
        const int n = nt * 16 + (lane & 15);
        for (int f = 0; f < 4; ++f) {
            const v16h bfr = *(const v16h*)(Wb + ((((size_t)f * 8 + nt) * 32 + lane) << 4));
            c = __builtin_amdgcn_wmma_f32_16x16x32_f16(false, a2[f], false, bfr,
                                                       (short)0, c, false, false);
        }
        float bib = bb[n];
#pragma unroll
        for (int j = 0; j < 8; ++j) {
            int rr = j + ((lane >> 4) << 3);
            float v = c[j] + bib;
            if (GEL2) v = gelu_f(v);
            smem[rr * DH + n] = v;
        }
    }
    __syncthreads();
    // ---- LayerNorm: lane pair (L, L+16) shares row m, 64 features each ----
    {
        const int half = lane >> 4;
        const float* sr = smem + m * DH + half * 64;
        float s = 0.f, ss = 0.f;
        for (int t = 0; t < 16; ++t) {
            float4 v = *(const float4*)(sr + t * 4);
            s  += v.x + v.y + v.z + v.w;
            ss += v.x * v.x + v.y * v.y + v.z * v.z + v.w * v.w;
        }
        s  += __shfl_xor(s, 16, 32);
        ss += __shfl_xor(ss, 16, 32);
        float mu  = s * (1.0f / 128.0f);
        float var = ss * (1.0f / 128.0f) - mu * mu;
        float inv = rsqrtf(var + LNEPS);
        float* o1 = out1 + (size_t)row * DH + half * 64;
        float* o2 = out2 ? out2 + (size_t)row * DH + half * 64 : nullptr;
        const float* gm = gma + half * 64;
        const float* bt = bet + half * 64;
        for (int t = 0; t < 16; ++t) {
            float4 v = *(const float4*)(sr + t * 4);
            float4 o;
            o.x = (v.x - mu) * inv * gm[t * 4 + 0] + bt[t * 4 + 0];
            o.y = (v.y - mu) * inv * gm[t * 4 + 1] + bt[t * 4 + 1];
            o.z = (v.z - mu) * inv * gm[t * 4 + 2] + bt[t * 4 + 2];
            o.w = (v.w - mu) * inv * gm[t * 4 + 3] + bt[t * 4 + 3];
            *(float4*)(o1 + t * 4) = o;
            if (o2) *(float4*)(o2 + t * 4) = o;
        }
    }
}

// ---------------- APPNP: acc = selfw*h ; acc[dst]+=w*h[src] ; h' = .9acc+.1x
__global__ void k_prop_init(const float* __restrict__ ping,
                            const float* __restrict__ selfw, float* acc) {
    int t = blockIdx.x * blockDim.x + threadIdx.x;
    if (t >= NN * 32) return;
    int n = t >> 5, c = (t & 31) * 4;
    float sw = selfw[n];
    float4 hv = *(const float4*)(ping + (size_t)n * DH + c);
    float4 o = {sw * hv.x, sw * hv.y, sw * hv.z, sw * hv.w};
    *(float4*)(acc + (size_t)n * DH + c) = o;
}
__global__ void k_prop_edge(const int* __restrict__ ei, const float* __restrict__ dis,
                            const float* __restrict__ ping, float* acc) {
    int t = blockIdx.x * blockDim.x + threadIdx.x;
    if (t >= EE * 32) return;
    int e = t >> 5, c = (t & 31) * 4;
    if ((threadIdx.x & 31) == 0 && e + 4096 < EE)
        __builtin_prefetch(ei + e + 4096, 0, 0);   // stream edge indices (global_prefetch_b8)
    int s = ei[e], d = ei[EE + e];
    float w = dis[s] * dis[d];
    float4 hv = *(const float4*)(ping + (size_t)s * DH + c);
    float* ap = acc + (size_t)d * DH + c;
    atomicAdd(ap + 0, w * hv.x); atomicAdd(ap + 1, w * hv.y);
    atomicAdd(ap + 2, w * hv.z); atomicAdd(ap + 3, w * hv.w);
}
__global__ void k_prop_comb(const float* __restrict__ acc,
                            const float* __restrict__ x0, float* out) {
    int t = blockIdx.x * blockDim.x + threadIdx.x;
    if (t >= NN * 32) return;
    int n = t >> 5, c = (t & 31) * 4;
    float4 av = *(const float4*)(acc + (size_t)n * DH + c);
    float4 xv = *(const float4*)(x0 + (size_t)n * DH + c);
    float4 o = {(1.0f - ALPHA) * av.x + ALPHA * xv.x,
                (1.0f - ALPHA) * av.y + ALPHA * xv.y,
                (1.0f - ALPHA) * av.z + ALPHA * xv.z,
                (1.0f - ALPHA) * av.w + ALPHA * xv.w};
    *(float4*)(out + (size_t)n * DH + c) = o;
}

// ---------------- gate MLP (128->64->1) per node ----------------
__global__ void k_gate(const float* __restrict__ h, const float* __restrict__ Wg1,
                       const float* __restrict__ bg1, const float* __restrict__ Wg2,
                       const float* __restrict__ bg2, float* gate) {
    int n = blockIdx.x * blockDim.x + threadIdx.x;
    if (n >= NN) return;
    const float* hr = h + (size_t)n * DH;
    float acc[64];
#pragma unroll
    for (int j = 0; j < 64; ++j) acc[j] = 0.f;
    for (int f = 0; f < DH; ++f) {
        float hv = hr[f];
        const float* wr = Wg1 + f * 64;
#pragma unroll
        for (int j = 0; j < 64; ++j) acc[j] = fmaf(hv, wr[j], acc[j]);
    }
    float s = bg2[0];
#pragma unroll
    for (int j = 0; j < 64; ++j) s += gelu_f(acc[j] + bg1[j]) * Wg2[j];
    gate[n] = s;
}

// ---------------- attentional aggregation (segment softmax) ----------------
__global__ void k_segmax(const float* __restrict__ gate, const int* __restrict__ batch,
                         unsigned* gmax) {
    int n = blockIdx.x * blockDim.x + threadIdx.x;
    if (n >= NN) return;
    atomicMax(&gmax[batch[n]], fenc(gate[n]));
}
__global__ void k_segexp(float* gate, const int* __restrict__ batch,
                         const unsigned* __restrict__ gmax, float* gsum) {
    int n = blockIdx.x * blockDim.x + threadIdx.x;
    if (n >= NN) return;
    int b = batch[n];
    float ev = expf(gate[n] - fdec(gmax[b]));
    gate[n] = ev;
    atomicAdd(&gsum[b], ev);
}
__global__ void k_gagg(const float* __restrict__ h, const float* __restrict__ gate,
                       const int* __restrict__ batch, const float* __restrict__ gsum,
                       float* gout) {
    int t = blockIdx.x * blockDim.x + threadIdx.x;
    if (t >= NN * 32) return;
    int n = t >> 5, c = (t & 31) * 4;
    int b = batch[n];
    float a = gate[n] / gsum[b];
    float4 hv = *(const float4*)(h + (size_t)n * DH + c);
    float* gp = gout + (size_t)b * DH + c;
    atomicAdd(gp + 0, a * hv.x); atomicAdd(gp + 1, a * hv.y);
    atomicAdd(gp + 2, a * hv.z); atomicAdd(gp + 3, a * hv.w);
}

// ---------------- host orchestration ----------------
extern "C" void kernel_launch(void* const* d_in, const int* in_sizes, int n_in,
                              void* d_out, int out_size, void* d_ws, size_t ws_size,
                              hipStream_t stream) {
    const float* x    = (const float*)d_in[0];
    const int*   ei   = (const int*)d_in[1];      // edge_index [2,E]
    const int*   batc = (const int*)d_in[2];
    const float* ea   = (const float*)d_in[3];
    const float* We1 = (const float*)d_in[4];  const float* be1 = (const float*)d_in[5];
    const float* W1a = (const float*)d_in[6];  const float* b1a = (const float*)d_in[7];
    const float* W1b = (const float*)d_in[8];  const float* b1b = (const float*)d_in[9];
    const float* g1  = (const float*)d_in[10]; const float* bn1 = (const float*)d_in[11];
    const float* We2 = (const float*)d_in[12]; const float* be2 = (const float*)d_in[13];
    const float* W2a = (const float*)d_in[14]; const float* b2a = (const float*)d_in[15];
    const float* W2b = (const float*)d_in[16]; const float* b2b = (const float*)d_in[17];
    const float* g2  = (const float*)d_in[18]; const float* bn2 = (const float*)d_in[19];
    const float* Wg1 = (const float*)d_in[20]; const float* bg1 = (const float*)d_in[21];
    const float* Wg2 = (const float*)d_in[22]; const float* bg2 = (const float*)d_in[23];

    // workspace layout (~155 MB + 115 KB packed weights)
    float* ws   = (float*)d_ws;
    float* agg  = ws;                             // N*128 (scatter acc, reused)
    float* h1   = agg + (size_t)NN * DH;          // N*128 (conv1 out / APPNP ping)
    float* xA   = h1 + (size_t)NN * DH;           // N*128 (conv2 out = APPNP x)
    float* dis  = xA + (size_t)NN * DH;           // N  (rsqrt(deg+1))
    float* selfw = dis + NN;                      // N  (1/(deg+1))
    float* gate  = selfw + NN;                    // N
    unsigned* gmax = (unsigned*)(gate + NN);      // B
    float* gsum = (float*)(gmax + BB);            // B
    _Float16* pk1a = (_Float16*)(gsum + BB);      // 2*8*512  (32B-aligned offset)
    _Float16* pk1b = pk1a + 8192;                 // 4*8*512
    _Float16* pk2a = pk1b + 16384;
    _Float16* pk2b = pk2a + 16384;
    float* hout = (float*)d_out;                  // N*128
    float* gout = hout + (size_t)NN * DH;         // B*128

    auto nb = [](long n) { return (int)((n + 255) / 256); };

    // pre-pack weights into f16 WMMA-B fragment order (once per launch)
    k_pack<2><<<nb(8192),  256, 0, stream>>>(W1a, pk1a);
    k_pack<4><<<nb(16384), 256, 0, stream>>>(W1b, pk1b);
    k_pack<4><<<nb(16384), 256, 0, stream>>>(W2a, pk2a);
    k_pack<4><<<nb(16384), 256, 0, stream>>>(W2b, pk2b);

    // degrees / gcn norm
    k_zero<<<2048, 256, 0, stream>>>(dis, NN);
    k_deg<<<nb(EE), 256, 0, stream>>>(ei, dis);
    k_degfin<<<nb(NN), 256, 0, stream>>>(dis, selfw);

    // conv1: GINE(x) -> MLP(64->128->128) -> gelu -> LN -> h1
    k_zero<<<2048, 256, 0, stream>>>(agg, NN * DIN);
    k_gine_edge<DIN><<<nb((long)EE * 16), 256, 0, stream>>>(ei, x, ea, We1, be1, agg);
    k_mlp<DIN, true><<<NN / 16, 32, 0, stream>>>(x, agg, pk1a, b1a, pk1b, b1b,
                                                 g1, bn1, h1, nullptr);
    // conv2: GINE(h1) -> MLP(128->128->128) -> LN -> xA (and ping copy in h1)
    k_zero<<<2048, 256, 0, stream>>>(agg, NN * DH);
    k_gine_edge<DH><<<nb((long)EE * 32), 256, 0, stream>>>(ei, h1, ea, We2, be2, agg);
    k_mlp<DH, false><<<NN / 16, 32, 0, stream>>>(h1, agg, pk2a, b2a, pk2b, b2b,
                                                 g2, bn2, xA, h1);
    // APPNP, K=16 (ping=h1, acc=agg, x=xA); last combine writes d_out
    for (int k = 0; k < KITER; ++k) {
        k_prop_init<<<nb((long)NN * 32), 256, 0, stream>>>(h1, selfw, agg);
        k_prop_edge<<<nb((long)EE * 32), 256, 0, stream>>>(ei, dis, h1, agg);
        k_prop_comb<<<nb((long)NN * 32), 256, 0, stream>>>(
            agg, xA, (k == KITER - 1) ? hout : h1);
    }
    // attentional readout
    k_gate<<<nb(NN), 256, 0, stream>>>(hout, Wg1, bg1, Wg2, bg2, gate);
    k_zero<<<1, 256, 0, stream>>>((float*)gmax, 2 * BB);   // gmax(uint)+gsum, 0 == enc(<-inf)
    k_zero<<<BB, 256, 0, stream>>>(gout, BB * DH);
    k_segmax<<<nb(NN), 256, 0, stream>>>(gate, batc, gmax);
    k_segexp<<<nb(NN), 256, 0, stream>>>(gate, batc, gmax, gsum);
    k_gagg<<<nb((long)NN * 32), 256, 0, stream>>>(hout, gate, batc, gsum, gout);
}
